// HybridGCN_9491877724909
// MI455X (gfx1250) — matrix operand order repaired and checked by reading, not compiled, not run on hardware
//
#include <hip/hip_runtime.h>
#include <hip/hip_bf16.h>

// ---------------- problem constants ----------------
#define NN    100000
#define MPAD  100032    // NN padded to multiple of 64
#define EE    1600000
#define GG    512
#define DIN   770
#define KP0   832       // DIN padded to multiple of 64
#define HH    256
#define HA    128       // attention hidden

// ---------------- WMMA types ----------------
typedef __attribute__((ext_vector_type(16))) __bf16 v16bf;
typedef __attribute__((ext_vector_type(8)))  float  v8f;

union Frag { uint4 u[2]; v16bf v; };

__device__ __forceinline__ unsigned short f2bf(float f) {
  unsigned u = __float_as_uint(f);
  unsigned r = (u + 0x7FFFu + ((u >> 16) & 1u)) >> 16;  // round-nearest-even
  return (unsigned short)r;
}

__device__ __forceinline__ unsigned int enc_f(float f) {
  unsigned u = __float_as_uint(f);
  return (u & 0x80000000u) ? ~u : (u | 0x80000000u);
}
__device__ __forceinline__ float dec_f(unsigned int u) {
  return __uint_as_float((u & 0x80000000u) ? (u & 0x7FFFFFFFu) : ~u);
}

// ---------------- weight convert (f32 -> bf16, K-padded) ----------------
__global__ void cvt_pad_kernel(const float* __restrict__ Wf, unsigned short* __restrict__ Wb,
                               int K, int Kpad, int Nc) {
  int idx = blockIdx.x * blockDim.x + threadIdx.x;
  if (idx >= Kpad * Nc) return;
  int k = idx / Nc, n = idx % Nc;
  float v = (k < K) ? Wf[k * Nc + n] : 0.f;
  Wb[idx] = f2bf(v);
}

// x[N x 770] f32 -> xbf[MPAD x 832] bf16 (zero padded both dims); one block per row
__global__ __launch_bounds__(256) void cvt_x_kernel(const float* __restrict__ x,
                                                    unsigned short* __restrict__ xbf) {
  const int i = blockIdx.x;
  const bool rowOK = (i < NN);
  for (int k = threadIdx.x; k < KP0; k += 256) {
    float v = (rowOK && k < DIN) ? x[(long long)i * DIN + k] : 0.f;
    xbf[(long long)i * KP0 + k] = f2bf(v);
  }
}

// relu h in place + produce bf16 copy hbf[MPAD x 256]; one block per padded row
__global__ __launch_bounds__(256) void relu_cvt_kernel(float* __restrict__ h,
                                                       unsigned short* __restrict__ hbf) {
  const int i = blockIdx.x, f = threadIdx.x;
  unsigned short o = 0;
  if (i < NN) {
    const float v = fmaxf(h[(long long)i * HH + f], 0.f);
    h[(long long)i * HH + f] = v;
    o = f2bf(v);
  }
  hbf[(long long)i * HH + f] = o;
}

// ---------------- degree / rsqrt ----------------
__global__ void deg_kernel(const int* __restrict__ dst, float* __restrict__ deg, int E) {
  int e = blockIdx.x * blockDim.x + threadIdx.x;
  if (e < E) atomicAdd(&deg[dst[e]], 1.0f);
}
__global__ void rsqrt_kernel(float* __restrict__ deg, int N) {
  int i = blockIdx.x * blockDim.x + threadIdx.x;
  if (i < N) deg[i] = rsqrtf(1.0f + deg[i]);
}

// ---------------- bf16 WMMA GEMM: C[MPAD,Nc] = A[MPAD,Kpad]_bf16 * B[Kpad,Nc]_bf16 ------
// Fully padded operands: no bounds checks, no conversions, no divergence anywhere.
// block = 256 threads = 8 wave32; 64x64 output tile; K-tile = 64 (2 wmma substeps).
// Per stage, each wave issues 4 WMMAs (2 accumulators x 2 k-substeps) with A reuse.
__global__ __launch_bounds__(256) void gemm_bf16_wmma(
    const unsigned short* __restrict__ A, const unsigned short* __restrict__ B,
    float* __restrict__ C, int Kpad, int Nc)
{
  __shared__ __align__(16) unsigned short As[64][72];   // 64 m x 64 k (+8 pad)
  __shared__ __align__(16) unsigned short BsT[64][72];  // 64 n x 64 k (+8 pad)

  const int tid  = threadIdx.x;
  const int wave = tid >> 5;
  const int lane = tid & 31;
  const int m0 = blockIdx.y * 64;
  const int nb = blockIdx.x * 64;
  const int wm = wave & 3;       // 0..3 -> M sub-tile (16 rows)
  const int wn = wave >> 2;      // 0..1 -> N sub-tile (32 cols)

  v8f acc0 = {0.f, 0.f, 0.f, 0.f, 0.f, 0.f, 0.f, 0.f};
  v8f acc1 = {0.f, 0.f, 0.f, 0.f, 0.f, 0.f, 0.f, 0.f};

  const int srow = tid >> 2;          // 0..63 (A: m, B: k)
  const int sc16 = (tid & 3) * 16;    // 0,16,32,48 (A: k, B: n)

  const unsigned short* aBase = A + (long long)(m0 + srow) * Kpad + sc16;
  const unsigned short* bBase = B + (long long)srow * Nc + nb + sc16;

  for (int k0 = 0; k0 < Kpad; k0 += 64) {
    // stage A: straight row-major copy, 2x b128 load + 2x b128 LDS store
    {
      const uint4 qa0 = *(const uint4*)(aBase + k0);
      const uint4 qa1 = *(const uint4*)(aBase + k0 + 8);
      *(uint4*)&As[srow][sc16]     = qa0;
      *(uint4*)&As[srow][sc16 + 8] = qa1;
    }
    // stage B: transpose scatter (16 n x 1 k per thread)
    {
      const unsigned short* bp = bBase + (long long)k0 * Nc;
      const uint4 q0 = *(const uint4*)(bp);
      const uint4 q1 = *(const uint4*)(bp + 8);
      BsT[sc16 +  0][srow] = (unsigned short)(q0.x & 0xFFFFu);
      BsT[sc16 +  1][srow] = (unsigned short)(q0.x >> 16);
      BsT[sc16 +  2][srow] = (unsigned short)(q0.y & 0xFFFFu);
      BsT[sc16 +  3][srow] = (unsigned short)(q0.y >> 16);
      BsT[sc16 +  4][srow] = (unsigned short)(q0.z & 0xFFFFu);
      BsT[sc16 +  5][srow] = (unsigned short)(q0.z >> 16);
      BsT[sc16 +  6][srow] = (unsigned short)(q0.w & 0xFFFFu);
      BsT[sc16 +  7][srow] = (unsigned short)(q0.w >> 16);
      BsT[sc16 +  8][srow] = (unsigned short)(q1.x & 0xFFFFu);
      BsT[sc16 +  9][srow] = (unsigned short)(q1.x >> 16);
      BsT[sc16 + 10][srow] = (unsigned short)(q1.y & 0xFFFFu);
      BsT[sc16 + 11][srow] = (unsigned short)(q1.y >> 16);
      BsT[sc16 + 12][srow] = (unsigned short)(q1.z & 0xFFFFu);
      BsT[sc16 + 13][srow] = (unsigned short)(q1.z >> 16);
      BsT[sc16 + 14][srow] = (unsigned short)(q1.w & 0xFFFFu);
      BsT[sc16 + 15][srow] = (unsigned short)(q1.w >> 16);
    }
    __syncthreads();

    // A fragments (lanes 0-15 row=lane K 0..15; lanes 16-31 row=lane-16 K 16..31)
    Frag a0, a1, b00, b01, b10, b11;
    const unsigned short* ap = &As[wm * 16 + (lane & 15)][(lane >> 4) * 16];
    a0.u[0] = *(const uint4*)(ap);
    a0.u[1] = *(const uint4*)(ap + 8);
    a1.u[0] = *(const uint4*)(ap + 32);
    a1.u[1] = *(const uint4*)(ap + 40);
    // B fragments (lanes 0-15 col=lane K 0..15; lanes 16-31 col=lane-16 K 16..31)
    const unsigned short* bp0 = &BsT[wn * 32 + (lane & 15)][(lane >> 4) * 16];
    b00.u[0] = *(const uint4*)(bp0);
    b00.u[1] = *(const uint4*)(bp0 + 8);
    b01.u[0] = *(const uint4*)(bp0 + 32);
    b01.u[1] = *(const uint4*)(bp0 + 40);
    const unsigned short* bp1 = bp0 + 16 * 72;   // +16 columns
    b10.u[0] = *(const uint4*)(bp1);
    b10.u[1] = *(const uint4*)(bp1 + 8);
    b11.u[0] = *(const uint4*)(bp1 + 32);
    b11.u[1] = *(const uint4*)(bp1 + 40);

    acc0 = __builtin_amdgcn_wmma_f32_16x16x32_bf16(false, a0.v, false, b00.v,
                                                   (short)0, acc0, false, false);
    acc1 = __builtin_amdgcn_wmma_f32_16x16x32_bf16(false, a0.v, false, b10.v,
                                                   (short)0, acc1, false, false);
    acc0 = __builtin_amdgcn_wmma_f32_16x16x32_bf16(false, a1.v, false, b01.v,
                                                   (short)0, acc0, false, false);
    acc1 = __builtin_amdgcn_wmma_f32_16x16x32_bf16(false, a1.v, false, b11.v,
                                                   (short)0, acc1, false, false);
    __syncthreads();
  }

  // C/D layout: VGPR r: lanes 0-15 -> M=r, lanes 16-31 -> M=8+r (unconditional, C padded)
  const int n0 = nb + wn * 32 + (lane & 15);
  const long long mb = m0 + wm * 16 + ((lane >> 4) ? 8 : 0);
  #pragma unroll
  for (int r = 0; r < 8; ++r) {
    C[(mb + r) * Nc + n0]      = acc0[r];
    C[(mb + r) * Nc + n0 + 16] = acc1[r];
  }
}

// ---------------- GCN aggregation ----------------
// h[i,:] = t[i,:]*dis[i]^2 + b   (self loop + bias, pre-relu)
__global__ __launch_bounds__(256) void self_init_kernel(
    const float* __restrict__ t, const float* __restrict__ dis,
    const float* __restrict__ b, float* __restrict__ h) {
  const int i = blockIdx.x, f = threadIdx.x;
  const float d = dis[i];
  h[(long long)i * HH + f] = t[(long long)i * HH + f] * d * d + b[f];
}

// one wave per edge; lane covers 8 features (2x float4 gather, 8 f32 atomics)
__global__ __launch_bounds__(256) void edge_scatter_kernel(
    const int* __restrict__ src, const int* __restrict__ dst,
    const float* __restrict__ dis, const float* __restrict__ t,
    float* __restrict__ h, int E)
{
  const int e = (blockIdx.x * blockDim.x + threadIdx.x) >> 5;
  if (e >= E) return;
  const int lane = threadIdx.x & 31;
  const int s = src[e], d = dst[e];
  const float nrm = dis[s] * dis[d];
  const float* tr = t + (long long)s * HH + lane * 8;
  __builtin_prefetch(tr, 0, 0);                     // global_prefetch_b8
  const float4 v0 = *(const float4*)(tr);
  const float4 v1 = *(const float4*)(tr + 4);
  float* hp = h + (long long)d * HH + lane * 8;
  atomicAdd(hp + 0, v0.x * nrm); atomicAdd(hp + 1, v0.y * nrm);
  atomicAdd(hp + 2, v0.z * nrm); atomicAdd(hp + 3, v0.w * nrm);
  atomicAdd(hp + 4, v1.x * nrm); atomicAdd(hp + 5, v1.y * nrm);
  atomicAdd(hp + 6, v1.z * nrm); atomicAdd(hp + 7, v1.w * nrm);
}

// ---------------- attention pooling ----------------
// one wave per node: score = tanh(s1+ba1) . Wa2 + ba2 ; segment max
__global__ __launch_bounds__(256) void attn_score_kernel(
    const float* __restrict__ s1, const float* __restrict__ ba1,
    const float* __restrict__ Wa2, const float* __restrict__ ba2,
    const int* __restrict__ batch, float* __restrict__ scores,
    unsigned int* __restrict__ gmax, int N)
{
  const int node = (blockIdx.x * blockDim.x + threadIdx.x) >> 5;
  if (node >= N) return;
  const int lane = threadIdx.x & 31;
  float p = 0.f;
  #pragma unroll
  for (int j = 0; j < 4; ++j) {
    const int c = lane * 4 + j;
    p += tanhf(s1[(long long)node * HA + c] + ba1[c]) * Wa2[c];
  }
  #pragma unroll
  for (int off = 16; off > 0; off >>= 1) p += __shfl_xor(p, off, 32);
  if (lane == 0) {
    const float sc = p + ba2[0];
    scores[node] = sc;
    atomicMax(&gmax[batch[node]], enc_f(sc));
  }
}

__global__ void expsum_kernel(float* __restrict__ scores, const int* __restrict__ batch,
                              const unsigned int* __restrict__ gmax,
                              float* __restrict__ esum, int N) {
  int i = blockIdx.x * blockDim.x + threadIdx.x;
  if (i >= N) return;
  const int g = batch[i];
  const float e = __expf(scores[i] - dec_f(gmax[g]));
  scores[i] = e;
  atomicAdd(&esum[g], e);
}

// one wave per node: emb[g,:] += h[i,:] * (e_i / esum[g])
__global__ __launch_bounds__(256) void emb_kernel(
    const float* __restrict__ h, const float* __restrict__ e,
    const int* __restrict__ batch, const float* __restrict__ esum,
    float* __restrict__ emb, int N)
{
  const int node = (blockIdx.x * blockDim.x + threadIdx.x) >> 5;
  if (node >= N) return;
  const int lane = threadIdx.x & 31;
  const int g = batch[node];
  const float w = e[node] / esum[g];
  const float* hr = h + (long long)node * HH + lane * 8;
  const float4 v0 = *(const float4*)(hr);
  const float4 v1 = *(const float4*)(hr + 4);
  float* ep = emb + (long long)g * HH + lane * 8;
  atomicAdd(ep + 0, v0.x * w); atomicAdd(ep + 1, v0.y * w);
  atomicAdd(ep + 2, v0.z * w); atomicAdd(ep + 3, v0.w * w);
  atomicAdd(ep + 4, v1.x * w); atomicAdd(ep + 5, v1.y * w);
  atomicAdd(ep + 6, v1.z * w); atomicAdd(ep + 7, v1.w * w);
}

// ---------------- classifier: one block per graph ----------------
__global__ __launch_bounds__(128) void classifier_kernel(
    const float* __restrict__ emb, const float* __restrict__ Wc1,
    const float* __restrict__ bc1, const float* __restrict__ Wc2,
    const float* __restrict__ bc2, float* __restrict__ out)
{
  __shared__ float red[128];
  const int g = blockIdx.x, j = threadIdx.x;
  float a = bc1[j];
  for (int k = 0; k < HH; ++k) a += emb[(long long)g * HH + k] * Wc1[k * HA + j];
  red[j] = fmaxf(a, 0.f) * Wc2[j];
  __syncthreads();
  for (int s = 64; s > 0; s >>= 1) {
    if (j < s) red[j] += red[j + s];
    __syncthreads();
  }
  if (j == 0) out[g] = red[0] + bc2[0];
}

// ---------------- host launch ----------------
extern "C" void kernel_launch(void* const* d_in, const int* in_sizes, int n_in,
                              void* d_out, int out_size, void* d_ws, size_t ws_size,
                              hipStream_t stream) {
  (void)in_sizes; (void)n_in; (void)out_size; (void)ws_size;
  const float* x    = (const float*)d_in[0];
  const int*   ei   = (const int*)  d_in[1];
  const int*   batch= (const int*)  d_in[2];
  const float* W0 = (const float*)d_in[3];  const float* b0 = (const float*)d_in[4];
  const float* W1 = (const float*)d_in[5];  const float* b1 = (const float*)d_in[6];
  const float* W2 = (const float*)d_in[7];  const float* b2 = (const float*)d_in[8];
  const float* Wa1= (const float*)d_in[9];  const float* ba1= (const float*)d_in[10];
  const float* Wa2= (const float*)d_in[11]; const float* ba2= (const float*)d_in[12];
  const float* Wc1= (const float*)d_in[13]; const float* bc1= (const float*)d_in[14];
  const float* Wc2= (const float*)d_in[15]; const float* bc2= (const float*)d_in[16];
  float* out = (float*)d_out;
  const int* src = ei;
  const int* dst = ei + EE;

  // workspace carve-out (256B aligned chunks)
  char* ws = (char*)d_ws;
  size_t off = 0;
  auto carve = [&](size_t bytes) -> char* {
    char* p = ws + off; off += (bytes + 255) & ~(size_t)255; return p;
  };
  float*          dis  = (float*)carve((size_t)NN * 4);
  unsigned short* xbf  = (unsigned short*)carve((size_t)MPAD * KP0 * 2);
  unsigned short* hbf  = (unsigned short*)carve((size_t)MPAD * HH * 2);
  float*          t    = (float*)carve((size_t)MPAD * HH * 4);   // pre-agg / s1 (padded)
  float*          h    = (float*)carve((size_t)NN * HH * 4);     // f32 activations
  unsigned short* W0b  = (unsigned short*)carve((size_t)KP0 * HH * 2);
  unsigned short* W1b  = (unsigned short*)carve((size_t)HH * HH * 2);
  unsigned short* W2b  = (unsigned short*)carve((size_t)HH * HH * 2);
  unsigned short* Wa1b = (unsigned short*)carve((size_t)HH * HA * 2);
  float*          scr  = (float*)carve((size_t)NN * 4);
  unsigned int*   gmax = (unsigned int*)carve((size_t)GG * 4);
  float*          esum = (float*)carve((size_t)GG * 4);
  float*          emb  = (float*)carve((size_t)GG * HH * 4);

  hipMemsetAsync(dis,  0, (size_t)NN * 4, stream);
  hipMemsetAsync(gmax, 0, (size_t)GG * 4, stream);            // enc(-inf) floor
  hipMemsetAsync(esum, 0, (size_t)GG * 4, stream);
  hipMemsetAsync(emb,  0, (size_t)GG * HH * 4, stream);

  // weights -> bf16 (W0 K-padded 770 -> 832); x -> padded bf16
  cvt_pad_kernel<<<(KP0 * HH + 255) / 256, 256, 0, stream>>>(W0, W0b, DIN, KP0, HH);
  cvt_pad_kernel<<<(HH * HH + 255) / 256, 256, 0, stream>>>(W1, W1b, HH, HH, HH);
  cvt_pad_kernel<<<(HH * HH + 255) / 256, 256, 0, stream>>>(W2, W2b, HH, HH, HH);
  cvt_pad_kernel<<<(HH * HA + 255) / 256, 256, 0, stream>>>(Wa1, Wa1b, HH, HH, HA);
  cvt_x_kernel<<<MPAD, 256, 0, stream>>>(x, xbf);

  // symmetric normalization
  deg_kernel  <<<(EE + 255) / 256, 256, 0, stream>>>(dst, dis, EE);
  rsqrt_kernel<<<(NN + 255) / 256, 256, 0, stream>>>(dis, NN);

  const dim3 gH(HH / 64, MPAD / 64);
  const dim3 gA(HA / 64, MPAD / 64);
  const int edgeBlocks = (EE * 32 + 255) / 256;

  // layer 0: t = x @ W0 (WMMA bf16), then aggregate + relu(+bf16 cvt)
  gemm_bf16_wmma<<<gH, 256, 0, stream>>>(xbf, W0b, t, KP0, HH);
  self_init_kernel<<<NN, HH, 0, stream>>>(t, dis, b0, h);
  edge_scatter_kernel<<<edgeBlocks, 256, 0, stream>>>(src, dst, dis, t, h, EE);
  relu_cvt_kernel<<<MPAD, HH, 0, stream>>>(h, hbf);

  // layer 1
  gemm_bf16_wmma<<<gH, 256, 0, stream>>>(hbf, W1b, t, HH, HH);
  self_init_kernel<<<NN, HH, 0, stream>>>(t, dis, b1, h);
  edge_scatter_kernel<<<edgeBlocks, 256, 0, stream>>>(src, dst, dis, t, h, EE);
  relu_cvt_kernel<<<MPAD, HH, 0, stream>>>(h, hbf);

  // layer 2
  gemm_bf16_wmma<<<gH, 256, 0, stream>>>(hbf, W2b, t, HH, HH);
  self_init_kernel<<<NN, HH, 0, stream>>>(t, dis, b2, h);
  edge_scatter_kernel<<<edgeBlocks, 256, 0, stream>>>(src, dst, dis, t, h, EE);
  relu_cvt_kernel<<<MPAD, HH, 0, stream>>>(h, hbf);

  // attention pooling: s1 = h @ Wa1 (WMMA, writes t), score/softmax/emb
  gemm_bf16_wmma<<<gA, 256, 0, stream>>>(hbf, Wa1b, t, HH, HA);
  attn_score_kernel<<<(NN * 32 + 255) / 256, 256, 0, stream>>>(t, ba1, Wa2, ba2,
                                                               batch, scr, gmax, NN);
  expsum_kernel<<<(NN + 255) / 256, 256, 0, stream>>>(scr, batch, gmax, esum, NN);
  emb_kernel<<<(NN * 32 + 255) / 256, 256, 0, stream>>>(h, scr, batch, esum, emb, NN);

  // classifier
  classifier_kernel<<<GG, HA, 0, stream>>>(emb, Wc1, bc1, Wc2, bc2, out);
}
